// MultiHeadSelfAttention_36962488549722
// MI455X (gfx1250) — matrix operand order repaired and checked by reading, not compiled, not run on hardware
//
#include <hip/hip_runtime.h>
#include <hip/hip_bf16.h>
#include <math.h>

// ---------------- CDNA5 WMMA plumbing ----------------
typedef _Float16 h8  __attribute__((ext_vector_type(8)));
typedef _Float16 h16 __attribute__((ext_vector_type(16)));
typedef float    f8  __attribute__((ext_vector_type(8)));

#define B_  2
#define T_  2048
#define C_  1024
#define H_  16
#define D_  64
#define M_  (B_*T_)          // 4096 rows of x

__device__ __forceinline__ h16 cat88(h8 lo, h8 hi) {
  return __builtin_shufflevector(lo, hi, 0,1,2,3,4,5,6,7,8,9,10,11,12,13,14,15);
}

__device__ __forceinline__ f8 wmma_f16(h16 a, h16 b, f8 c) {
  // D = A(16x32 f16) * B(32x16 f16) + C(16x16 f32)
  return __builtin_amdgcn_wmma_f32_16x16x32_f16(false, a, false, b, (short)0, c, false, false);
}

// A fragment (16x32): rows m0..m0+15, k in [kb,kb+32), A row-major with leading dim lda.
// Contiguous k order (load_bt uses the same): lane L holds row L&15; lanes 0-15 get K {kb..kb+15},
// lanes 16-31 get K {kb+16..kb+31}.  -> two aligned 16B loads.
__device__ __forceinline__ h16 load_a(const _Float16* A, int lda, int m0, int kb, int lane) {
  const _Float16* p = A + (size_t)(m0 + (lane & 15)) * lda + kb + ((lane & 16) ? 16 : 0);
  h8 lo = *(const h8*)(p);
  h8 hi = *(const h8*)(p + 8);
  return cat88(lo, hi);
}

// B fragment (32x16) from Bt stored N-major: Bt[n*ldk + k].
// Contiguous k order (same as load_a): lane L holds col L&15; K = (L<16 ? 0 : 16) + e, contiguous in k. -> 16 contiguous halves.
__device__ __forceinline__ h16 load_bt(const _Float16* Bt, int ldk, int n0, int kb, int lane) {
  const _Float16* p = Bt + (size_t)(n0 + (lane & 15)) * ldk + kb + ((lane & 16) ? 16 : 0);
  h8 lo = *(const h8*)(p);
  h8 hi = *(const h8*)(p + 8);
  return cat88(lo, hi);
}
// C/D fragment element r maps to (row = r + 8*(lane>=16), col = lane&15).

// ---------------- conversion kernels ----------------
__global__ void cvt_f32_f16(const float* __restrict__ s, _Float16* __restrict__ d, int n) {
  int i = blockIdx.x * blockDim.x + threadIdx.x;
  if (i < n) d[i] = (_Float16)s[i];
}

// W is (K x N) row-major fp32; emit WT as (N x K) f16 so GEMM B-fragments are contiguous.
__global__ void transpose_cvt(const float* __restrict__ W, _Float16* __restrict__ WT, int K, int N) {
  int i = blockIdx.x * blockDim.x + threadIdx.x;
  if (i < K * N) {
    int k = i / N, n = i - k * N;
    WT[(size_t)n * K + k] = (_Float16)W[i];
  }
}

// ---------------- fused QKV GEMM + bias + RoPE + pack ----------------
// One wave computes a 32x64 tile of qkv = Xh @ Wqkv + b.  N-tile width 64 == one head's
// slab of q, k or v, so RoPE pairs (d, d+32) sit in the same lane, fragments nb and nb^2.
__global__ __launch_bounds__(256) void qkv_rope_wmma(
    const _Float16* __restrict__ Xh, const _Float16* __restrict__ WT,
    const float* __restrict__ bias,
    _Float16* __restrict__ qh, _Float16* __restrict__ kh, _Float16* __restrict__ vt)
{
  const int lane = threadIdx.x & 31;
  const int tile = blockIdx.x * 8 + (threadIdx.x >> 5);
  const int TN = (3 * C_) / 64;                 // 48 column tiles
  const int tm = tile / TN, tn = tile - tm * TN;
  const int m0 = tm * 32, n0 = tn * 64;

  f8 acc[2][4] = {};
  for (int kb = 0; kb < C_; kb += 32) {
    h16 a0 = load_a(Xh, C_, m0,      kb, lane);
    h16 a1 = load_a(Xh, C_, m0 + 16, kb, lane);
#pragma unroll
    for (int nb = 0; nb < 4; ++nb) {
      h16 bfrag = load_bt(WT, C_, n0 + 16 * nb, kb, lane);
      acc[0][nb] = wmma_f16(a0, bfrag, acc[0][nb]);
      acc[1][nb] = wmma_f16(a1, bfrag, acc[1][nb]);
    }
  }

  const int region = n0 >> 10;                  // 0=q 1=k 2=v
  const int head   = (n0 & (C_ - 1)) >> 6;
  const int col    = lane & 15;
  const int roff   = (lane & 16) ? 8 : 0;

#pragma unroll
  for (int mi = 0; mi < 2; ++mi) {
#pragma unroll
    for (int r = 0; r < 8; ++r) {
      const int row = m0 + 16 * mi + roff + r;
      const int bb  = row >> 11;                // row / T_
      const int t   = row & (T_ - 1);
      float v[4];
#pragma unroll
      for (int nb = 0; nb < 4; ++nb)
        v[nb] = acc[mi][nb][r] + bias[n0 + 16 * nb + col];

      if (region == 2) {
        // v: store transposed (B,H,D,T) so PV B-fragments are k-contiguous
#pragma unroll
        for (int nb = 0; nb < 4; ++nb) {
          const int d = 16 * nb + col;
          vt[(((size_t)(bb * H_ + head)) * D_ + d) * T_ + t] = (_Float16)v[nb];
        }
      } else {
        _Float16* dst = (region == 0) ? qh : kh;
        const float scale = (region == 0) ? 0.125f : 1.0f;   // fold 1/sqrt(D) into q
#pragma unroll
        for (int nb = 0; nb < 4; ++nb) {
          const int d = 16 * nb + col;
          const int i = d & 31;
          const float ang = (float)t * __powf(10000.0f, -(float)i * (1.0f / 32.0f));
          float sn, cs;
          __sincosf(ang, &sn, &cs);
          const float pair = v[nb ^ 2];          // partner d +/- 32, same lane
          const float out  = (d < 32) ? (v[nb] * cs - pair * sn)
                                      : (v[nb] * cs + pair * sn);
          dst[(((size_t)(bb * H_ + head)) * T_ + t) * D_ + d] = (_Float16)(out * scale);
        }
      }
    }
  }
}

// ---------------- causal flash attention, one wave per (b,h,16-row q tile) ----------------
__global__ __launch_bounds__(32) void attn_flash_wmma(
    const _Float16* __restrict__ qh, const _Float16* __restrict__ kh,
    const _Float16* __restrict__ vt, _Float16* __restrict__ yh)
{
  __shared__ __align__(16) _Float16 plds[16 * 16];
  const int lane = threadIdx.x & 31;
  const int bh = blockIdx.x >> 7;               // b*H + h
  const int q0 = (blockIdx.x & 127) * 16;
  const _Float16* Q = qh + (size_t)bh * T_ * D_;
  const _Float16* K = kh + (size_t)bh * T_ * D_;
  const _Float16* V = vt + (size_t)bh * D_ * T_;

  const h16 aq0 = load_a(Q, D_, q0, 0,  lane);  // Q tile 16x64 stays in registers
  const h16 aq1 = load_a(Q, D_, q0, 32, lane);

  const int col  = lane & 15;
  const int roff = (lane & 16) ? 8 : 0;

  float mrow[8], lrow[8];
#pragma unroll
  for (int r = 0; r < 8; ++r) { mrow[r] = -INFINITY; lrow[r] = 0.0f; }
  f8 o[4] = {};

  for (int s0 = 0; s0 <= q0; s0 += 16) {
    f8 s = {};
    s = wmma_f16(aq0, load_bt(K, D_, s0, 0,  lane), s);
    s = wmma_f16(aq1, load_bt(K, D_, s0, 32, lane), s);

    if (s0 == q0) {                              // diagonal block: mask s>t
#pragma unroll
      for (int r = 0; r < 8; ++r)
        if (col > roff + r) s[r] = -INFINITY;
    }

#pragma unroll
    for (int r = 0; r < 8; ++r) {
      float mx = s[r];
#pragma unroll
      for (int off = 1; off < 16; off <<= 1)     // xor 1,2,4,8 stays within lane half
        mx = fmaxf(mx, __shfl_xor(mx, off, 32));
      const float newm  = fmaxf(mrow[r], mx);
      const float alpha = __expf(mrow[r] - newm);
      mrow[r] = newm;
      const float p = __expf(s[r] - newm);
      float ps = p;
#pragma unroll
      for (int off = 1; off < 16; off <<= 1)
        ps += __shfl_xor(ps, off, 32);
      lrow[r] = lrow[r] * alpha + ps;
      plds[(roff + r) * 16 + col] = (_Float16)p;  // stage P for C->A relayout
#pragma unroll
      for (int nb = 0; nb < 4; ++nb)
        o[nb][r] = o[nb][r] * alpha;
    }
    __syncthreads();

    h16 ap;                                       // P as A-fragment, K 16..31 zeroed
    {
      const _Float16* pp = plds + (lane & 15) * 16;
      h8 lo = *(const h8*)pp;
      h8 hi = *(const h8*)(pp + 8);
      const bool up = (lane & 16) != 0;
#pragma unroll
      for (int i = 0; i < 8; ++i) { ap[i] = up ? (_Float16)0.0f : lo[i]; ap[i + 8] = up ? (_Float16)0.0f : hi[i]; }
    }

#pragma unroll
    for (int nb = 0; nb < 4; ++nb) {
      // All lanes read V[d][s0..s0+15]; the K>=16 half is multiplied by A's zeros.
      const _Float16* p = V + (size_t)(16 * nb + col) * T_ + s0;
      h8 lo = *(const h8*)(p);
      h8 hi = *(const h8*)(p + 8);
      o[nb] = wmma_f16(ap, cat88(lo, hi), o[nb]);
    }
    __syncthreads();
  }

  const int bb = bh >> 4, head = bh & 15;
#pragma unroll
  for (int nb = 0; nb < 4; ++nb)
#pragma unroll
    for (int r = 0; r < 8; ++r) {
      const int t = q0 + roff + r;
      const int d = 16 * nb + col;
      yh[((size_t)(bb * T_ + t)) * C_ + head * D_ + d] = (_Float16)(o[nb][r] / lrow[r]);
    }
}

// ---------------- generic WMMA GEMM with fp32 + bias epilogue (out projection) ----------------
__global__ __launch_bounds__(256) void gemm_bias_wmma(
    const _Float16* __restrict__ A, const _Float16* __restrict__ Bt,
    const float* __restrict__ bias, float* __restrict__ Cout,
    int M, int N, int Kd)
{
  const int lane = threadIdx.x & 31;
  const int tile = blockIdx.x * 8 + (threadIdx.x >> 5);
  const int TN = N / 64;
  const int tm = tile / TN, tn = tile - tm * TN;
  const int m0 = tm * 32, n0 = tn * 64;

  f8 acc[2][4] = {};
  for (int kb = 0; kb < Kd; kb += 32) {
    h16 a0 = load_a(A, Kd, m0,      kb, lane);
    h16 a1 = load_a(A, Kd, m0 + 16, kb, lane);
#pragma unroll
    for (int nb = 0; nb < 4; ++nb) {
      h16 bfrag = load_bt(Bt, Kd, n0 + 16 * nb, kb, lane);
      acc[0][nb] = wmma_f16(a0, bfrag, acc[0][nb]);
      acc[1][nb] = wmma_f16(a1, bfrag, acc[1][nb]);
    }
  }
  const int col = lane & 15;
  const int roff = (lane & 16) ? 8 : 0;
#pragma unroll
  for (int mi = 0; mi < 2; ++mi)
#pragma unroll
    for (int r = 0; r < 8; ++r) {
      const int row = m0 + 16 * mi + roff + r;
#pragma unroll
      for (int nb = 0; nb < 4; ++nb) {
        const int n = n0 + 16 * nb + col;
        Cout[(size_t)row * N + n] = acc[mi][nb][r] + bias[n];
      }
    }
}

// ---------------- host-side launch ----------------
extern "C" void kernel_launch(void* const* d_in, const int* in_sizes, int n_in,
                              void* d_out, int out_size, void* d_ws, size_t ws_size,
                              hipStream_t stream)
{
  (void)in_sizes; (void)n_in; (void)out_size; (void)ws_size;
  const float* x     = (const float*)d_in[0];
  const float* Wqkv  = (const float*)d_in[1];
  const float* bqkv  = (const float*)d_in[2];
  const float* Wproj = (const float*)d_in[3];
  const float* bproj = (const float*)d_in[4];
  float* out = (float*)d_out;

  char* ws = (char*)d_ws;
  size_t off = 0;
  auto alloc = [&](size_t bytes) -> char* {
    char* p = ws + off;
    off += (bytes + 255) & ~(size_t)255;
    return p;
  };
  _Float16* xh  = (_Float16*)alloc((size_t)M_ * C_ * 2);        // 8 MB
  _Float16* WqT = (_Float16*)alloc((size_t)3 * C_ * C_ * 2);    // 6 MB
  _Float16* WpT = (_Float16*)alloc((size_t)C_ * C_ * 2);        // 2 MB
  _Float16* qh  = (_Float16*)alloc((size_t)B_ * H_ * T_ * D_ * 2);
  _Float16* kh  = (_Float16*)alloc((size_t)B_ * H_ * T_ * D_ * 2);
  _Float16* vt  = (_Float16*)alloc((size_t)B_ * H_ * T_ * D_ * 2);
  _Float16* yh  = (_Float16*)alloc((size_t)M_ * C_ * 2);        // ~48 MB total

  cvt_f32_f16<<<(M_ * C_ + 255) / 256, 256, 0, stream>>>(x, xh, M_ * C_);
  transpose_cvt<<<(C_ * 3 * C_ + 255) / 256, 256, 0, stream>>>(Wqkv, WqT, C_, 3 * C_);
  transpose_cvt<<<(C_ * C_ + 255) / 256, 256, 0, stream>>>(Wproj, WpT, C_, C_);

  // (M/32)*(3C/64) tiles, 8 waves per block
  qkv_rope_wmma<<<(M_ / 32) * ((3 * C_) / 64) / 8, 256, 0, stream>>>(xh, WqT, bqkv, qh, kh, vt);

  attn_flash_wmma<<<B_ * H_ * (T_ / 16), 32, 0, stream>>>(qh, kh, vt, yh);

  gemm_bias_wmma<<<(M_ / 32) * (C_ / 64) / 8, 256, 0, stream>>>(yh, WpT, bproj, out, M_, C_, C_);
}
